// RelDecoderLayer_19619410608621
// MI455X (gfx1250) — matrix operand order repaired and checked by reading, not compiled, not run on hardware
//
#include <hip/hip_runtime.h>

// ---------------------------------------------------------------------------
// Types for CDNA5 WMMA (wave32, 16x16x32 bf16 -> f32) and TDM descriptors
// ---------------------------------------------------------------------------
typedef __attribute__((ext_vector_type(16))) __bf16 bf16x16;
typedef __attribute__((ext_vector_type(8)))  float  f32x8;
typedef __attribute__((ext_vector_type(4)))  unsigned int u32x4;
typedef __attribute__((ext_vector_type(8)))  int    i32x8;
typedef __attribute__((ext_vector_type(4)))  int    i32x4;

union FragU { u32x4 q[2]; bf16x16 v; };

__device__ __forceinline__ unsigned short f2bf(float f) {
  unsigned u = __float_as_uint(f);
  u += 0x7FFFu + ((u >> 16) & 1u);          // round-to-nearest-even
  return (unsigned short)(u >> 16);
}

__device__ __forceinline__ float block_sum(float v, float* red) {
  const int lane = threadIdx.x & 31;
  const int w    = threadIdx.x >> 5;
  #pragma unroll
  for (int off = 16; off; off >>= 1) v += __shfl_xor(v, off, 32);
  if (lane == 0) red[w] = v;
  __syncthreads();
  float s = (threadIdx.x < 8) ? red[threadIdx.x] : 0.0f;
  if (w == 0) {
    #pragma unroll
    for (int off = 4; off; off >>= 1) s += __shfl_xor(s, off, 8);
    if (lane == 0) red[0] = s;
  }
  __syncthreads();
  float r = red[0];
  __syncthreads();                           // allow reuse of red[]
  return r;
}

// ---------------------------------------------------------------------------
// Tensor Data Mover: 2-D tile (bf16) global -> LDS, per ISA 08_async_tensor §8.
// group0: count=1 | lds_addr | global_addr(57b) | type=2
// group1: data_size=1(2B) | tensor_dim0/1 | tile_dim0/1 | tensor_dim0_stride
// remaining groups: zero (tile is 2-D). Issued by one wave; tracked TENSORcnt.
// This toolchain exposes the 6-arg builtin (clang-23 / therock-10.0 form).
// ---------------------------------------------------------------------------
__device__ __forceinline__ void tdm_load_2d_bf16(unsigned lds_addr, const void* gaddr,
                                                 unsigned td0, unsigned td1,
                                                 unsigned tile0, unsigned tile1,
                                                 unsigned long long stride0) {
  const unsigned long long ga = (unsigned long long)(size_t)gaddr;
  u32x4 g0;
  g0[0] = 1u;                                        // count=1 (valid user D#)
  g0[1] = lds_addr;                                  // LDS byte address
  g0[2] = (unsigned)ga;                              // global_addr[31:0]
  g0[3] = (unsigned)((ga >> 32) & 0x01FFFFFFull)     // global_addr[56:32]
        | (2u << 30);                                // type=2 ("image")
  i32x8 g1;
  g1[0] = (int)(1u << 16);                           // data_size=1 (2 bytes)
  g1[1] = (int)((td0 & 0xFFFFu) << 16);              // tensor_dim0[15:0]
  g1[2] = (int)((td0 >> 16) | ((td1 & 0xFFFFu) << 16));    // d0[31:16] | d1[15:0]
  g1[3] = (int)((td1 >> 16) | ((tile0 & 0xFFFFu) << 16));  // d1[31:16] | tile_dim0
  g1[4] = (int)(tile1 & 0xFFFFu);                    // tile_dim1 (tile_dim2 = 0)
  g1[5] = (int)(stride0 & 0xFFFFFFFFull);            // tensor_dim0_stride[31:0]
  g1[6] = (int)((stride0 >> 32) & 0xFFFFull);        // stride[47:32] (dim1_stride=0)
  g1[7] = 0;
  i32x4 gz4; gz4[0] = 0; gz4[1] = 0; gz4[2] = 0; gz4[3] = 0;
  i32x8 gz8;
  #pragma unroll
  for (int i = 0; i < 8; ++i) gz8[i] = 0;
  __builtin_amdgcn_tensor_load_to_lds(g0, g1, gz4, gz4, gz8, 0);
}

// ---------------------------------------------------------------------------
// Elementwise cast f32 -> bf16
// ---------------------------------------------------------------------------
__global__ __launch_bounds__(256)
void cast_bf16_kernel(const float* __restrict__ in, unsigned short* __restrict__ out, size_t n) {
  for (size_t i = (size_t)blockIdx.x * 256 + threadIdx.x; i < n; i += (size_t)gridDim.x * 256)
    out[i] = f2bf(in[i]);
}

// ---------------------------------------------------------------------------
// Transpose + cast: in f32 [N][K] -> out bf16 [K][N]   (grid: (K/32, N/32))
// ---------------------------------------------------------------------------
__global__ __launch_bounds__(256)
void transpose_cast_kernel(const float* __restrict__ in, unsigned short* __restrict__ out,
                           int N, int K) {
  __shared__ float tile[32][33];
  const int n0 = blockIdx.y * 32, k0 = blockIdx.x * 32;
  const int tx = threadIdx.x & 31, ty = threadIdx.x >> 5;   // ty 0..7
  #pragma unroll
  for (int r = ty; r < 32; r += 8)
    tile[r][tx] = in[(size_t)(n0 + r) * K + k0 + tx];
  __syncthreads();
  #pragma unroll
  for (int r = ty; r < 32; r += 8)
    out[(size_t)(k0 + r) * N + n0 + tx] = f2bf(tile[tx][r]);
}

// ---------------------------------------------------------------------------
// Tiled bf16 WMMA GEMM: C[M][N] f32 = A[M][K] bf16 (row-major) * B[K][N] bf16
// 256 threads = 8 waves (4x2), tile 128x128x32, TDM double-buffered LDS:
// wave 0 issues tensor_load_to_lds for tile t+1 while all waves run WMMA on
// tile t; synchronization via s_wait_tensorcnt + workgroup barrier.
// ---------------------------------------------------------------------------
__global__ __launch_bounds__(256)
void gemm_bf16_wmma(const unsigned short* __restrict__ A,
                    const unsigned short* __restrict__ B,
                    float* __restrict__ C, int M, int N, int K) {
  constexpr int TBM = 128, TBN = 128, TBK = 32;
  __shared__ unsigned short As[2][TBM][TBK];   // 8KB x2, tile rows contiguous
  __shared__ unsigned short Bs[2][TBK][TBN];   // 8KB x2, tile rows contiguous
  const int tid  = threadIdx.x;
  const int lane = tid & 31;
  const int wave = tid >> 5;
  const int wm   = wave >> 1;                  // 0..3 -> 32-row slab
  const int wn   = wave & 1;                   // 0..1 -> 64-col slab
  const int m0   = blockIdx.y * TBM;
  const int n0   = blockIdx.x * TBN;

  // LDS byte addresses (flat->LDS: low 32 bits of generic pointer)
  const unsigned ldsA[2] = { (unsigned)(size_t)&As[0][0][0], (unsigned)(size_t)&As[1][0][0] };
  const unsigned ldsB[2] = { (unsigned)(size_t)&Bs[0][0][0], (unsigned)(size_t)&Bs[1][0][0] };

  f32x8 acc[2][4];
  #pragma unroll
  for (int i = 0; i < 2; ++i)
    #pragma unroll
    for (int j = 0; j < 4; ++j)
      #pragma unroll
      for (int v = 0; v < 8; ++v) acc[i][j][v] = 0.0f;

  const int nk = K / TBK;
  // Prologue: tile 0 via TDM into buffer 0
  if (wave == 0) {
    tdm_load_2d_bf16(ldsA[0], A + (size_t)m0 * K, (unsigned)K, (unsigned)M,
                     TBK, TBM, (unsigned long long)K);
    tdm_load_2d_bf16(ldsB[0], B + n0, (unsigned)N, (unsigned)K,
                     TBN, TBK, (unsigned long long)N);
  }
  __builtin_amdgcn_s_wait_tensorcnt(0);
  __syncthreads();

  int buf = 0;
  for (int t = 0; t < nk; ++t) {
    if (wave == 0 && t + 1 < nk) {             // DMA next tile into back buffer
      const int k0 = (t + 1) * TBK;
      tdm_load_2d_bf16(ldsA[buf ^ 1], A + (size_t)m0 * K + k0, (unsigned)K, (unsigned)M,
                       TBK, TBM, (unsigned long long)K);
      tdm_load_2d_bf16(ldsB[buf ^ 1], B + (size_t)k0 * N + n0, (unsigned)N, (unsigned)K,
                       TBN, TBK, (unsigned long long)N);
    }
    // A fragments: lane row = lane%16; lanes 0-15 K {0..7,16..23}, 16-31 K {8..15,24..31}
    const int khalf = (lane >> 4) << 3;
    const int arow  = lane & 15;
    bf16x16 afrag[2], bfrag[4];
    #pragma unroll
    for (int mf = 0; mf < 2; ++mf) {
      FragU f;
      const unsigned short* p = &As[buf][wm * 32 + mf * 16 + arow][0];
      f.q[0] = *reinterpret_cast<const u32x4*>(p + khalf);
      f.q[1] = *reinterpret_cast<const u32x4*>(p + 16 + khalf);
      afrag[mf] = f.v;
    }
    // B fragments: lane col n = lane%16; K order same as A fragments
    #pragma unroll
    for (int nf = 0; nf < 4; ++nf) {
      FragU f;
      const unsigned short* p = &Bs[buf][0][wn * 64 + nf * 16 + arow];
      #pragma unroll
      for (int e = 0; e < 16; ++e)
        reinterpret_cast<unsigned short*>(&f)[e] = p[(khalf + (e & 7) + ((e >> 3) << 4)) * TBN];
      bfrag[nf] = f.v;
    }
    #pragma unroll
    for (int mf = 0; mf < 2; ++mf)
      #pragma unroll
      for (int nf = 0; nf < 4; ++nf)
        acc[mf][nf] = __builtin_amdgcn_wmma_f32_16x16x32_bf16(
            false, afrag[mf], false, bfrag[nf], (short)0, acc[mf][nf], false, false);
    __syncthreads();                           // all waves done reading buf
    __builtin_amdgcn_s_wait_tensorcnt(0);      // wave 0: back buffer landed
    __syncthreads();                           // publish back buffer
    buf ^= 1;
  }
  // C layout: lane<16: n=lane, m=v ; lane>=16: n=lane-16, m=v+8
  const int cn = lane & 15;
  const int mh = (lane >> 4) << 3;
  #pragma unroll
  for (int mf = 0; mf < 2; ++mf)
    #pragma unroll
    for (int nf = 0; nf < 4; ++nf)
      #pragma unroll
      for (int v = 0; v < 8; ++v) {
        const int mm = m0 + wm * 32 + mf * 16 + mh + v;
        const int nn = n0 + wn * 64 + nf * 16 + cn;
        C[(size_t)mm * N + nn] = acc[mf][nf][v];
      }
}

// ---------------------------------------------------------------------------
// Pack Q/K/V f32 [i*B+b][h*64+d] into per-(b,h) bf16 tensors:
//   Qp,Vp: [(b*16+h)*2048 + i][64]    Kt: [(b*16+h)*64 + d][2048]
// ---------------------------------------------------------------------------
__global__ __launch_bounds__(256)
void pack_attn_kernel(const float* __restrict__ Qf, const float* __restrict__ Kf,
                      const float* __restrict__ Vf,
                      unsigned short* __restrict__ Qp, unsigned short* __restrict__ Kt,
                      unsigned short* __restrict__ Vp) {
  const size_t idx = (size_t)blockIdx.x * 256 + threadIdx.x;   // < 4194304
  const int r = (int)(idx >> 10), c = (int)(idx & 1023);
  const int i = r >> 1, b = r & 1, h = c >> 6, d = c & 63;
  const size_t bh = (size_t)(b * 16 + h);
  Qp[(bh * 2048 + i) * 64 + d] = f2bf(Qf[idx]);
  Kt[(bh * 64 + d) * 2048 + i] = f2bf(Kf[idx]);
  Vp[(bh * 2048 + i) * 64 + d] = f2bf(Vf[idx]);
}

// ---------------------------------------------------------------------------
// Flash attention (causal), bf16 WMMA. grid (S/128, NH, B), 8 waves/WG,
// one 16-query-row tile per wave. Output avec bf16 [i*B+b][h*64+d].
// ---------------------------------------------------------------------------
__global__ __launch_bounds__(256)
void flash_attn_kernel(const unsigned short* __restrict__ Qp,
                       const unsigned short* __restrict__ Kt,
                       const unsigned short* __restrict__ Vp,
                       unsigned short* __restrict__ avec) {
  __shared__ unsigned short Pb[8][16][32];     // per-wave P tile (bf16)
  const int tid = threadIdx.x, lane = tid & 31, wave = tid >> 5;
  const int h = blockIdx.y, b = blockIdx.z;
  const int i0 = blockIdx.x * 128 + wave * 16;
  const size_t baseQ = (size_t)(b * 16 + h) * 2048 * 64;
  const size_t baseK = (size_t)(b * 16 + h) * 64 * 2048;

  const int arow  = lane & 15;
  const int khalf = (lane >> 4) << 3;
  const int mh    = (lane >> 4) << 3;
  const int cn    = lane & 15;

  bf16x16 q[2];
  #pragma unroll
  for (int kk = 0; kk < 2; ++kk) {
    FragU f;
    const unsigned short* p = Qp + baseQ + (size_t)(i0 + arow) * 64 + kk * 32;
    f.q[0] = *reinterpret_cast<const u32x4*>(p + khalf);
    f.q[1] = *reinterpret_cast<const u32x4*>(p + 16 + khalf);
    q[kk] = f.v;
  }

  f32x8 o[4];
  #pragma unroll
  for (int d = 0; d < 4; ++d)
    #pragma unroll
    for (int v = 0; v < 8; ++v) o[d][v] = 0.0f;
  float rm[8], rs[8];
  #pragma unroll
  for (int v = 0; v < 8; ++v) { rm[v] = -1e30f; rs[v] = 0.0f; }

  const int iLast = i0 + 15;
  for (int j0 = 0; j0 <= iLast; j0 += 32) {    // wave-uniform loop: EXEC stays full
    // prefetch next j-tile of K^T / V while this tile computes
    __builtin_prefetch(Kt + baseK + (size_t)lane * 2048 + j0 + 32, 0, 1);
    __builtin_prefetch(Vp + baseQ + (size_t)(j0 + 32 + lane) * 64, 0, 1);
    f32x8 s[2];
    #pragma unroll
    for (int jf = 0; jf < 2; ++jf) {
      f32x8 c;
      #pragma unroll
      for (int v = 0; v < 8; ++v) c[v] = 0.0f;
      #pragma unroll
      for (int kk = 0; kk < 2; ++kk) {
        FragU f;  // B frag from Kt: lane j = lane%16, d in A's K order
        const unsigned short* p = Kt + baseK + (size_t)(kk * 32) * 2048 + j0 + jf * 16 + cn;
        #pragma unroll
        for (int e = 0; e < 16; ++e)
          reinterpret_cast<unsigned short*>(&f)[e] = p[(size_t)(khalf + (e & 7) + ((e >> 3) << 4)) * 2048];
        c = __builtin_amdgcn_wmma_f32_16x16x32_bf16(false, q[kk], false, f.v, (short)0, c, false, false);
      }
      s[jf] = c;
    }
    // scale + causal mask + row max over this 32-wide tile
    float tmax[8];
    #pragma unroll
    for (int v = 0; v < 8; ++v) {
      const int m = i0 + mh + v;
      float best = -1e30f;
      #pragma unroll
      for (int jf = 0; jf < 2; ++jf) {
        const int n = j0 + jf * 16 + cn;
        const float val = (n <= m) ? s[jf][v] * 0.125f : -1e30f;
        s[jf][v] = val;
        best = fmaxf(best, val);
      }
      float x = best;
      #pragma unroll
      for (int off = 8; off; off >>= 1) x = fmaxf(x, __shfl_xor(x, off, 16));
      tmax[v] = x;
    }
    // online softmax update
    #pragma unroll
    for (int v = 0; v < 8; ++v) {
      const float nm = fmaxf(rm[v], tmax[v]);
      const float alpha = __expf(rm[v] - nm);
      float ts = 0.0f;
      #pragma unroll
      for (int jf = 0; jf < 2; ++jf) {
        const float pe = (s[jf][v] > -1e29f) ? __expf(s[jf][v] - nm) : 0.0f;
        s[jf][v] = pe;
        ts += pe;
      }
      #pragma unroll
      for (int off = 8; off; off >>= 1) ts += __shfl_xor(ts, off, 16);
      rs[v] = rs[v] * alpha + ts;
      rm[v] = nm;
      #pragma unroll
      for (int d = 0; d < 4; ++d) o[d][v] *= alpha;
    }
    // P -> LDS (bf16), reload as 16x32 A fragment (same-wave LDS is in-order)
    #pragma unroll
    for (int jf = 0; jf < 2; ++jf)
      #pragma unroll
      for (int v = 0; v < 8; ++v)
        Pb[wave][mh + v][jf * 16 + cn] = f2bf(s[jf][v]);
    FragU pf;
    {
      const unsigned short* p = &Pb[wave][arow][0];
      pf.q[0] = *reinterpret_cast<const u32x4*>(p + khalf);
      pf.q[1] = *reinterpret_cast<const u32x4*>(p + 16 + khalf);
    }
    // O += P * V  (V rows are contiguous over d)
    #pragma unroll
    for (int d = 0; d < 4; ++d) {
      FragU vf;
      const unsigned short* p = Vp + baseQ + (size_t)j0 * 64 + d * 16 + cn;
      #pragma unroll
      for (int e = 0; e < 16; ++e)
        reinterpret_cast<unsigned short*>(&vf)[e] = p[(size_t)(khalf + (e & 7) + ((e >> 3) << 4)) * 64];
      o[d] = __builtin_amdgcn_wmma_f32_16x16x32_bf16(false, pf.v, false, vf.v, (short)0, o[d], false, false);
    }
  }
  // normalize, store bf16 to avec[(i*B+b)*1024 + h*64 + d]
  #pragma unroll
  for (int v = 0; v < 8; ++v) {
    const float inv = 1.0f / rs[v];
    const size_t row = (size_t)(i0 + mh + v) * 2 + b;
    #pragma unroll
    for (int d = 0; d < 4; ++d)
      avec[row * 1024 + h * 64 + d * 16 + cn] = f2bf(o[d][v] * inv);
  }
}

// ---------------------------------------------------------------------------
// h = LayerNorm(x + y); optional bf16 copy. One row (1024) per block.
// ---------------------------------------------------------------------------
__global__ __launch_bounds__(256)
void add_ln_kernel(const float* __restrict__ x, const float* __restrict__ y,
                   const float* __restrict__ gamma, const float* __restrict__ beta,
                   float* __restrict__ outf, unsigned short* __restrict__ outbf) {
  __shared__ float red[8];
  const size_t row = blockIdx.x;
  const float* xr = x + row * 1024;
  const float* yr = y + row * 1024;
  const int tid = threadIdx.x;
  float v[4];
  float s = 0.0f;
  #pragma unroll
  for (int e = 0; e < 4; ++e) { const int c = tid + 256 * e; v[e] = xr[c] + yr[c]; s += v[e]; }
  const float mu = block_sum(s, red) * (1.0f / 1024.0f);
  float vs = 0.0f;
  #pragma unroll
  for (int e = 0; e < 4; ++e) { const float d = v[e] - mu; vs += d * d; }
  const float var = block_sum(vs, red) * (1.0f / 1024.0f);
  const float inv = rsqrtf(var + 1e-5f);
  #pragma unroll
  for (int e = 0; e < 4; ++e) {
    const int c = tid + 256 * e;
    const float o = gamma[c] * (v[e] - mu) * inv + beta[c];
    outf[row * 1024 + c] = o;
    if (outbf) outbf[row * 1024 + c] = f2bf(o);
  }
}

// ---------------------------------------------------------------------------
// KL: two-stage deterministic reduction of (m^2 - 2s + exp(2s)) * scale
// ---------------------------------------------------------------------------
__global__ __launch_bounds__(256)
void kl_partial_kernel(const float* __restrict__ mean, const float* __restrict__ stdv,
                       int n, float scale, float* __restrict__ partial) {
  __shared__ float red[8];
  float s = 0.0f;
  for (int i = blockIdx.x * 256 + threadIdx.x; i < n; i += gridDim.x * 256) {
    const float m = mean[i], sd = stdv[i];
    s += m * m - 2.0f * sd + __expf(2.0f * sd);
  }
  s = block_sum(s, red);
  if (threadIdx.x == 0) partial[blockIdx.x] = s * scale;
}

__global__ __launch_bounds__(256)
void kl_final_kernel(const float* __restrict__ partial, float* __restrict__ dst) {
  __shared__ float red[8];
  float s = 0.0f;
  for (int i = threadIdx.x; i < 1536; i += 256) s += partial[i];
  s = block_sum(s, red);
  if (threadIdx.x == 0) dst[0] = s;
}

// ---------------------------------------------------------------------------
// Host orchestration
// ---------------------------------------------------------------------------
extern "C" void kernel_launch(void* const* d_in, const int* in_sizes, int n_in,
                              void* d_out, int out_size, void* d_ws, size_t ws_size,
                              hipStream_t stream) {
  (void)in_sizes; (void)n_in; (void)out_size; (void)ws_size;
  const float* dec = (const float*)d_in[0];
  // d_in[1] = attn_mask (causal; computed analytically, unused)
  const float* qm  = (const float*)d_in[2];
  const float* km  = (const float*)d_in[3];
  const float* vm  = (const float*)d_in[4];
  const float* qs  = (const float*)d_in[5];
  const float* ksd = (const float*)d_in[6];
  const float* vsd = (const float*)d_in[7];
  const float* om  = (const float*)d_in[8];
  const float* osd = (const float*)d_in[9];
  const float* g1  = (const float*)d_in[10];
  const float* b1  = (const float*)d_in[11];
  const float* w1  = (const float*)d_in[12];
  const float* w2  = (const float*)d_in[13];
  const float* w1s = (const float*)d_in[14];
  const float* w2s = (const float*)d_in[15];
  const float* g2  = (const float*)d_in[16];
  const float* b2  = (const float*)d_in[17];
  float* out = (float*)d_out;
  char* ws = (char*)d_ws;

  // ---- workspace layout (bytes), with phase-based aliasing (~166 MB) ----
  const size_t o_xbf  = 0;                        // X bf16        8 MB
  const size_t o_qmt  = o_xbf  + 8388608;         // qm^T bf16     2 MB
  const size_t o_kmt  = o_qmt  + 2097152;
  const size_t o_vmt  = o_kmt  + 2097152;
  const size_t o_w1t  = o_vmt  + 2097152;         // w1^T bf16     8 MB
  const size_t o_w2t  = o_w1t  + 8388608;         // w2^T bf16     8 MB
  const size_t o_part = o_w2t  + 8388608;         // KL partials   8 KB
  const size_t o_ra   = o_part + 8192;            // 32 MB region: Qp|Kt|Vp|avec, later l1 bf16
  const size_t o_qp   = o_ra;
  const size_t o_kt   = o_qp + 8388608;
  const size_t o_vp   = o_kt + 8388608;
  const size_t o_avec = o_vp + 8388608;
  const size_t o_l1bf = o_ra;                     // alias (Qp..avec dead)
  const size_t o_hf   = o_ra + 33554432;          // h f32        16 MB
  const size_t o_hbf  = o_hf + 16777216;          // h bf16        8 MB
  const size_t o_l2f  = o_hbf + 8388608;          // l2 f32       16 MB
  const size_t o_rb   = o_l2f + 16777216;         // 64 MB region: Qf/Kf/Vf -> aout -> l1 f32
  const size_t o_qf   = o_rb;
  const size_t o_kf   = o_rb + 16777216;
  const size_t o_vf   = o_rb + 33554432;
  const size_t o_aout = o_rb;                     // alias (Qf/Kf/Vf dead)
  const size_t o_l1f  = o_rb;                     // alias (aout dead)

  unsigned short* XBF  = (unsigned short*)(ws + o_xbf);
  unsigned short* QMT  = (unsigned short*)(ws + o_qmt);
  unsigned short* KMT  = (unsigned short*)(ws + o_kmt);
  unsigned short* VMT  = (unsigned short*)(ws + o_vmt);
  unsigned short* W1T  = (unsigned short*)(ws + o_w1t);
  unsigned short* W2T  = (unsigned short*)(ws + o_w2t);
  float*          PART = (float*)(ws + o_part);
  unsigned short* QP   = (unsigned short*)(ws + o_qp);
  unsigned short* KT   = (unsigned short*)(ws + o_kt);
  unsigned short* VP   = (unsigned short*)(ws + o_vp);
  unsigned short* AVEC = (unsigned short*)(ws + o_avec);
  unsigned short* L1BF = (unsigned short*)(ws + o_l1bf);
  float*          HF   = (float*)(ws + o_hf);
  unsigned short* HBF  = (unsigned short*)(ws + o_hbf);
  float*          L2F  = (float*)(ws + o_l2f);
  float*          QF   = (float*)(ws + o_qf);
  float*          KF   = (float*)(ws + o_kf);
  float*          VF   = (float*)(ws + o_vf);
  float*          AOUT = (float*)(ws + o_aout);
  float*          L1F  = (float*)(ws + o_l1f);

  // 1) casts + weight transposes
  cast_bf16_kernel<<<4096, 256, 0, stream>>>(dec, XBF, (size_t)4194304);
  transpose_cast_kernel<<<dim3(32, 32),  256, 0, stream>>>(qm, QMT, 1024, 1024);
  transpose_cast_kernel<<<dim3(32, 32),  256, 0, stream>>>(km, KMT, 1024, 1024);
  transpose_cast_kernel<<<dim3(32, 32),  256, 0, stream>>>(vm, VMT, 1024, 1024);
  transpose_cast_kernel<<<dim3(32, 128), 256, 0, stream>>>(w1, W1T, 4096, 1024);
  transpose_cast_kernel<<<dim3(128, 32), 256, 0, stream>>>(w2, W2T, 1024, 4096);

  // 2) QKV projections (WMMA + TDM)
  gemm_bf16_wmma<<<dim3(8, 32), 256, 0, stream>>>(XBF, QMT, QF, 4096, 1024, 1024);
  gemm_bf16_wmma<<<dim3(8, 32), 256, 0, stream>>>(XBF, KMT, KF, 4096, 1024, 1024);
  gemm_bf16_wmma<<<dim3(8, 32), 256, 0, stream>>>(XBF, VMT, VF, 4096, 1024, 1024);

  // 3) attention (flash, WMMA)
  pack_attn_kernel<<<16384, 256, 0, stream>>>(QF, KF, VF, QP, KT, VP);
  flash_attn_kernel<<<dim3(16, 16, 2), 256, 0, stream>>>(QP, KT, VP, AVEC);

  // 4) (bug-faithful) output projection with value_mean^T, then LN1
  gemm_bf16_wmma<<<dim3(8, 32), 256, 0, stream>>>(AVEC, VMT, AOUT, 4096, 1024, 1024);
  add_ln_kernel<<<4096, 256, 0, stream>>>(dec, AOUT, g1, b1, HF, HBF);

  // 5) FF (no activation), then LN2 -> final output
  gemm_bf16_wmma<<<dim3(32, 32), 256, 0, stream>>>(HBF, W1T, L1F, 4096, 4096, 1024);
  cast_bf16_kernel<<<16384, 256, 0, stream>>>(L1F, L1BF, (size_t)16777216);
  gemm_bf16_wmma<<<dim3(8, 32), 256, 0, stream>>>(L1BF, W2T, L2F, 4096, 1024, 4096);
  add_ln_kernel<<<4096, 256, 0, stream>>>(HF, L2F, g2, b2, out, (unsigned short*)nullptr);

  // 6) KL scalar -> out[S*B*D]
  const float cA = 1.0f / (2.0f * 4194304.0f);   // attn group: 4 x 1M elems
  const float cF = 1.0f / (2.0f * 8388608.0f);   // ff group:   2 x 4M elems
  kl_partial_kernel<<<256, 256, 0, stream>>>(qm, qs,  1048576, cA, PART + 0);
  kl_partial_kernel<<<256, 256, 0, stream>>>(km, ksd, 1048576, cA, PART + 256);
  kl_partial_kernel<<<256, 256, 0, stream>>>(vm, vsd, 1048576, cA, PART + 512);
  kl_partial_kernel<<<256, 256, 0, stream>>>(om, osd, 1048576, cA, PART + 768);
  kl_partial_kernel<<<256, 256, 0, stream>>>(w1, w1s, 4194304, cF, PART + 1024);
  kl_partial_kernel<<<256, 256, 0, stream>>>(w2, w2s, 4194304, cF, PART + 1280);
  kl_final_kernel<<<1, 256, 0, stream>>>(PART, out + 4194304);
}
